// GlobalPointer_44684839747947
// MI455X (gfx1250) — matrix operand order repaired and checked by reading, not compiled, not run on hardware
//
#include <hip/hip_runtime.h>

typedef __attribute__((ext_vector_type(16))) _Float16 v16h;
typedef __attribute__((ext_vector_type(8)))  _Float16 v8h;
typedef __attribute__((ext_vector_type(4)))  _Float16 v4h;
typedef __attribute__((ext_vector_type(8)))  float    v8f;

#define HEADS 12
#define HEADD 64
#define BSZ   16
#define SEQ   512
#define HID   1024
#define NOUT  1536              // HEADS * HEADD * 2
#define NEGV  1000000000000.0f

// ---------------------------------------------------------------- convert X -> f16
__global__ __launch_bounds__(256) void cvt_x_f16(const float* __restrict__ x,
                                                 _Float16* __restrict__ xh, int n) {
    int i = (blockIdx.x * blockDim.x + threadIdx.x) * 4;
    if (i < n) {
        float4 v = *(const float4*)(x + i);
        v4h o;
        o[0] = (_Float16)v.x; o[1] = (_Float16)v.y;
        o[2] = (_Float16)v.z; o[3] = (_Float16)v.w;
        *(v4h*)(xh + i) = o;
    }
}

// ---------------------------------------------- W (H x NOUT) -> Wt (NOUT x H) f16
__global__ __launch_bounds__(256) void cvt_w_t(const float* __restrict__ W,
                                               _Float16* __restrict__ Wt) {
    int t = blockIdx.x * blockDim.x + threadIdx.x;   // t < NOUT*HID
    int k = t & (HID - 1);
    int n = t >> 10;
    Wt[t] = (_Float16)W[(size_t)k * NOUT + n];
}

// --------------------------------------------------------- projection + bias + RoPE
// Each wave: 16(M) x 32(N) (two C tiles, A reused). Block = 8 waves -> 64x64 tile.
// Grid (NOUT/64, 8192/64).
__global__ __launch_bounds__(256) void proj_rope(const _Float16* __restrict__ Xh,
                                                 const _Float16* __restrict__ Wt,
                                                 const float* __restrict__ bias,
                                                 const float* __restrict__ pe,
                                                 _Float16* __restrict__ qo,
                                                 _Float16* __restrict__ ko) {
    const int lane = threadIdx.x & 31;
    const int wave = threadIdx.x >> 5;
    const int m0 = blockIdx.y * 64 + (wave >> 1) * 16;
    const int n0 = blockIdx.x * 64 + (wave & 1) * 32;
    const int grp = lane >> 4;      // half-wave id
    const int lid = lane & 15;

    const _Float16* arow  = Xh + (size_t)(m0 + lid) * HID + grp * 16;       // A: M row
    const _Float16* brow0 = Wt + (size_t)(n0 + lid) * HID + grp * 16;       // B col n0+lid
    const _Float16* brow1 = Wt + (size_t)(n0 + 16 + lid) * HID + grp * 16;  // B col n0+16+lid

    v8f acc[2] = {};
    for (int k0 = 0; k0 < HID; k0 += 32) {
        v8h alo = *(const v8h*)(arow + k0);        // K = k0 + 0..7   (+16 upper half-wave)
        v8h ahi = *(const v8h*)(arow + k0 + 8);    // K = k0 + 8..15  (+16 upper half-wave)
        v16h a;
#pragma unroll
        for (int i = 0; i < 8; ++i) { a[i] = alo[i]; a[i + 8] = ahi[i]; }
        v16h bm0 = *(const v16h*)(brow0 + k0);     // 16 contiguous K values per lane
        v16h bm1 = *(const v16h*)(brow1 + k0);
        acc[0] = __builtin_amdgcn_wmma_f32_16x16x32_f16(false, a, false, bm0,
                                                        (short)0, acc[0], false, false);
        acc[1] = __builtin_amdgcn_wmma_f32_16x16x32_f16(false, a, false, bm1,
                                                        (short)0, acc[1], false, false);
    }

    // epilogue: bias + RoPE, write q/k as (b, h, l, d) f16
#pragma unroll
    for (int t = 0; t < 2; ++t) {
        const int n  = n0 + t * 16 + lid;   // output column 0..1535
        const int h  = n >> 7;              // head
        const int c2 = n & 127;             // within-head col: [0,64)=q, [64,128)=k
        const int dd = c2 & 63;             // rotary dim
        _Float16* dst = (c2 & 64) ? ko : qo;
        const float bv = bias[n];
#pragma unroll
        for (int i = 0; i < 8; ++i) {
            const int m    = m0 + grp * 8 + i;   // C layout: lanes16-31 -> M = i+8
            const int bidx = m >> 9;
            const int l    = m & (SEQ - 1);
            float v  = acc[t][i] + bv;
            float pr = __shfl_xor(v, 1, 32);     // interleave partner (n ^ 1)
            float rot = (n & 1) ? pr : -pr;      // even d: -x[d+1]; odd d: x[d-1]
            const float* p = pe + l * HEADD + (dd & ~1);
            float sv = p[0], cv = p[1];          // sin = pe[...,0::2], cos = pe[...,1::2]
            float o = v * cv + rot * sv;
            dst[((size_t)((bidx * HEADS + h) * SEQ + l)) * HEADD + dd] = (_Float16)o;
        }
    }
}

// ----------------------------------------------- logits = Q K^T, mask, tril, scale
// Each wave: 16(M) x 32(N), Q operand reused. Grid (SEQ/16, SEQ/32/8, B*HEADS).
__global__ __launch_bounds__(256) void attn_logits(const _Float16* __restrict__ qo,
                                                   const _Float16* __restrict__ ko,
                                                   const int* __restrict__ mask,
                                                   float* __restrict__ out) {
    const int lane = threadIdx.x & 31;
    const int wave = threadIdx.x >> 5;
    const int bh = blockIdx.z;
    const int b  = bh / HEADS;
    const int m0 = blockIdx.x * 16;
    const int n0 = (blockIdx.y * 8 + wave) * 32;
    const int grp = lane >> 4;
    const int lid = lane & 15;

    const _Float16* qrow  = qo + ((size_t)bh * SEQ + m0 + lid) * HEADD + grp * 16;
    const _Float16* krow0 = ko + ((size_t)bh * SEQ + n0 + lid) * HEADD + grp * 16;
    const _Float16* krow1 = ko + ((size_t)bh * SEQ + n0 + 16 + lid) * HEADD + grp * 16;

    v8f acc[2] = {};
#pragma unroll
    for (int k0 = 0; k0 < HEADD; k0 += 32) {
        v8h alo = *(const v8h*)(qrow + k0);
        v8h ahi = *(const v8h*)(qrow + k0 + 8);
        v16h a;
#pragma unroll
        for (int i = 0; i < 8; ++i) { a[i] = alo[i]; a[i + 8] = ahi[i]; }
        v16h bm0 = *(const v16h*)(krow0 + k0);
        v16h bm1 = *(const v16h*)(krow1 + k0);
        acc[0] = __builtin_amdgcn_wmma_f32_16x16x32_f16(false, a, false, bm0,
                                                        (short)0, acc[0], false, false);
        acc[1] = __builtin_amdgcn_wmma_f32_16x16x32_f16(false, a, false, bm1,
                                                        (short)0, acc[1], false, false);
    }

    float* orow = out + (size_t)bh * SEQ * SEQ;
#pragma unroll
    for (int t = 0; t < 2; ++t) {
        const int n = n0 + t * 16 + lid;
        const float pm = (float)mask[b * SEQ + n];
#pragma unroll
        for (int i = 0; i < 8; ++i) {
            const int m = m0 + grp * 8 + i;
            float v = acc[t][i];
            v = v * pm - (1.0f - pm) * NEGV;        // padding mask (key side)
            v -= (m > n) ? NEGV : 0.0f;             // strict lower triangle
            // 201 MB, never re-read: stream past L2 (NT hint) to keep q/k resident
            __builtin_nontemporal_store(v * 0.125f, &orow[(size_t)m * SEQ + n]);
        }
    }
}

extern "C" void kernel_launch(void* const* d_in, const int* in_sizes, int n_in,
                              void* d_out, int out_size, void* d_ws, size_t ws_size,
                              hipStream_t stream) {
    const float* x    = (const float*)d_in[0];   // (16,512,1024) f32
    const int*   mask = (const int*)  d_in[1];   // (16,512) i32
    const float* W    = (const float*)d_in[2];   // (1024,1536) f32
    const float* bias = (const float*)d_in[3];   // (1536,) f32
    const float* pe   = (const float*)d_in[4];   // (512,64) f32
    float* out = (float*)d_out;                  // (16,12,512,512) f32

    char* ws = (char*)d_ws;                      // offsets 256B-aligned
    _Float16* Xh = (_Float16*)(ws + 0);          // 16 MB
    _Float16* Wt = (_Float16*)(ws + 16777216);   //  3 MB
    _Float16* qo = (_Float16*)(ws + 19922944);   // 12 MB
    _Float16* ko = (_Float16*)(ws + 32505856);   // 12 MB  (total ~43 MB)

    cvt_x_f16<<<8192, 256, 0, stream>>>(x, Xh, BSZ * SEQ * HID);
    cvt_w_t  <<<6144, 256, 0, stream>>>(W, Wt);
    proj_rope<<<dim3(NOUT / 64, (BSZ * SEQ) / 64), 256, 0, stream>>>(Xh, Wt, bias, pe, qo, ko);
    attn_logits<<<dim3(SEQ / 16, SEQ / 32 / 8, BSZ * HEADS), 256, 0, stream>>>(qo, ko, mask, out);
}
